// DecoderBlock_85727547228743
// MI455X (gfx1250) — matrix operand order repaired and checked by reading, not compiled, not run on hardware
//
#include <hip/hip_runtime.h>

// ---------------------------------------------------------------------------
// Decoder block (self-attn + cross-attn + FFN) for gfx1250 (MI455X).
// GEMMs: v_wmma_f32_16x16x32_bf16 (wave32) with 128x64 weight tiles staged in
// LDS by the Tensor Data Mover (tensor_load_to_lds + s_wait_tensorcnt),
// double-buffered; activations stream via global_load_b128 + prefetch.
// ---------------------------------------------------------------------------

typedef __bf16 bf16_t;
typedef __bf16 v16bf __attribute__((ext_vector_type(16)));
typedef __bf16 v8bf  __attribute__((ext_vector_type(8)));
typedef float  v8f   __attribute__((ext_vector_type(8)));
typedef int    v4i_  __attribute__((ext_vector_type(4)));
typedef int    v8i_  __attribute__((ext_vector_type(8)));

#define WMMA_BF16(a, b, c) \
  __builtin_amdgcn_wmma_f32_16x16x32_bf16(false, (a), false, (b), (short)0, (c), false, false)

#define SCALE_QK 0.044194173824159216f  /* 512^-0.5 */

// TDM-staged 128(n) x 64(k) bf16 weight tile in LDS: 128B data rows + 16B TDM
// pad (pad_interval = 32 dwords, pad_amount = 4 dwords) => 144B pitch = 72
// elems. Bank stride 36 dwords -> 16 distinct banks across lanes; all
// fragment chunks stay 16B aligned.
#define BTILE_PITCH 72

// A-operand fragment: rows m0..m0+15, K k0..k0+31 of row-major bf16 [*][ld].
// Lane l holds row (l&15); lanes 0-15 K=k0..k0+15, lanes 16-31 K=k0+16..k0+31.
__device__ __forceinline__ v16bf load_a_frag(const bf16_t* __restrict__ base,
                                             int row0, int k0, int ld) {
  int l = threadIdx.x & 31;
  const bf16_t* p = base + (size_t)(row0 + (l & 15)) * ld + k0 + ((l >> 4) << 4);
  v8bf lo = *(const v8bf*)p;
  v8bf hi = *(const v8bf*)(p + 8);
  return __builtin_shufflevector(lo, hi, 0,1,2,3,4,5,6,7,8,9,10,11,12,13,14,15);
}

// B-operand fragment from a row-major [n][k] array in global memory.
// Lane l holds col (l&15); lanes 0-15 K=k0..k0+15, lanes 16-31 K=k0+16..k0+31.
__device__ __forceinline__ v16bf load_b_frag(const bf16_t* __restrict__ base,
                                             int n0, int k0, int ld) {
  int l = threadIdx.x & 31;
  const bf16_t* p = base + (size_t)(n0 + (l & 15)) * ld + k0 + ((l >> 4) << 4);
  v8bf lo = *(const v8bf*)p;
  v8bf hi = *(const v8bf*)(p + 8);
  return __builtin_shufflevector(lo, hi, 0,1,2,3,4,5,6,7,8,9,10,11,12,13,14,15);
}

// B-operand fragment from the TDM-staged LDS tile [128][BTILE_PITCH].
// ks selects the 32-wide k-substep (0 or 1) inside the 64-wide tile row.
__device__ __forceinline__ v16bf load_b_frag_lds(const bf16_t* base, int n0, int ks) {
  int l = threadIdx.x & 31;
  const bf16_t* p = base + (n0 + (l & 15)) * BTILE_PITCH + ks * 32 + ((l >> 4) << 4);
  v8bf lo = *(const v8bf*)p;       // ds_load_b128
  v8bf hi = *(const v8bf*)(p + 8); // ds_load_b128
  return __builtin_shufflevector(lo, hi, 0,1,2,3,4,5,6,7,8,9,10,11,12,13,14,15);
}

// ---------------------------------------------------------------------------
// Tensor Data Mover: DMA a 128(n) x 64(k) bf16 tile of Bt (row-major [n][K])
// starting at gptr into LDS at lds_addr, 16B pad after each 128B row.
// D# per CDNA5 ISA ch.8: group0 (4 SGPRs) + group1 (8 SGPRs), 2D tile form.
__device__ __forceinline__ void tdm_load_b_tile(unsigned lds_addr,
                                                const bf16_t* gptr, int K) {
  unsigned long long ga = (unsigned long long)(uintptr_t)gptr;
  v4i_ g0;
  g0[0] = 1;                                   // count=1 (valid user D#)
  g0[1] = (int)lds_addr;                       // lds_addr (bytes)
  g0[2] = (int)(unsigned)ga;                   // global_addr[31:0]
  g0[3] = (int)((unsigned)((ga >> 32) & 0x01FFFFFFu) | (2u << 30)); // hi + type=2
  unsigned td0 = (unsigned)K;                  // tensor_dim0 (k extent, OOB bound)
  unsigned td1 = 128u;                         // tensor_dim1 (rows)
  v8i_ g1;
  g1[0] = (int)((1u << 16)                     // data_size = 2 bytes
              | (1u << 20)                     // pad_enable
              | (4u << 22)                     // pad_interval: 32 dwords (128B)
              | (3u << 25));                   // pad_amount: 4 dwords (16B)
  g1[1] = (int)((td0 & 0xFFFFu) << 16);        // tensor_dim0[15:0]
  g1[2] = (int)((td0 >> 16) | ((td1 & 0xFFFFu) << 16)); // dim0 hi | dim1 lo
  g1[3] = (int)((td1 >> 16) | (64u << 16));    // dim1 hi | tile_dim0 = 64
  g1[4] = (int)128u;                           // tile_dim1 = 128, tile_dim2 = 0
  g1[5] = (int)td0;                            // tensor_dim0_stride[31:0] = K
  g1[6] = 0;                                   // stride hi / dim1_stride lo
  g1[7] = 0;                                   // dim1_stride hi
  asm volatile("tensor_load_to_lds %0, %1" :: "s"(g0), "s"(g1) : "memory");
}

// ---------------------------------------------------------------------------
// Weight transpose + fp32->bf16:  Wt[n][k] = (bf16)W[k][n]
__global__ __launch_bounds__(256) void wtrans_kernel(const float* __restrict__ W,
                                                     bf16_t* __restrict__ Wt,
                                                     int K, int N) {
  int idx = blockIdx.x * 256 + threadIdx.x;  // over N*K
  int n = idx / K;
  int k = idx - n * K;
  Wt[idx] = (bf16_t)W[(size_t)k * N + n];
}

// Elementwise fp32 -> bf16
__global__ __launch_bounds__(256) void cvt_kernel(const float* __restrict__ in,
                                                  bf16_t* __restrict__ out, int n) {
  int idx = blockIdx.x * 256 + threadIdx.x;
  if (idx < n) out[idx] = (bf16_t)in[idx];
}

// ---------------------------------------------------------------------------
// LayerNorm over last dim (512), wave-per-row, bf16 output.
__global__ __launch_bounds__(256) void ln_kernel(const float* __restrict__ x,
                                                 const float* __restrict__ g,
                                                 const float* __restrict__ be,
                                                 bf16_t* __restrict__ out) {
  int row = blockIdx.x * 8 + (threadIdx.x >> 5);
  int l = threadIdx.x & 31;
  const float* xr = x + (size_t)row * 512;
  float v[16];
  float s = 0.f, ss = 0.f;
#pragma unroll
  for (int i = 0; i < 16; ++i) {
    float t = xr[i * 32 + l];
    v[i] = t; s += t; ss += t * t;
  }
#pragma unroll
  for (int off = 16; off; off >>= 1) {
    s  += __shfl_xor(s,  off, 32);
    ss += __shfl_xor(ss, off, 32);
  }
  float mean = s * (1.f / 512.f);
  float var  = ss * (1.f / 512.f) - mean * mean;
  float rstd = rsqrtf(var + 1e-5f);
  bf16_t* orow = out + (size_t)row * 512;
#pragma unroll
  for (int i = 0; i < 16; ++i) {
    int c = i * 32 + l;
    orow[c] = (bf16_t)((v[i] - mean) * rstd * g[c] + be[c]);
  }
}

// ---------------------------------------------------------------------------
// Generic bf16 GEMM: C[M,N] = A[M,K] (row-major bf16) @ Bt[N,K]^T (bf16).
// Workgroup tile 128x128; 8 waves; each wave 32x64 = 2x4 WMMA tiles.
// 128x64 weight tile staged in LDS by the TDM, double-buffered; wave 0
// issues the DMA and synchronizes on TENSORcnt, barrier publishes the tile.
// 16 WMMAs per wave per barrier epoch.
// Epilogue: optional bias, relu, fp32 residual add, fp32/bf16 stores,
// and "transV" scatter into vt[b,h,d,t] (m=b*256+t, n=h*64+d).
__global__ __launch_bounds__(256) void gemm_bf16_kernel(
    const bf16_t* __restrict__ A, const bf16_t* __restrict__ Bt,
    int M, int N, int K,
    const float* __restrict__ bias, const float* __restrict__ resid,
    float* __restrict__ outF, bf16_t* __restrict__ outB,
    int relu, int transV) {
  __shared__ __align__(16) bf16_t smemB[2][128 * BTILE_PITCH];

  int w = threadIdx.x >> 5;
  int l = threadIdx.x & 31;
  int m0   = blockIdx.y * 128 + (w & 3) * 32;
  int nblk = blockIdx.x * 128;
  int n0l  = (w >> 2) * 64;   // local n inside the 128-wide tile

  const bf16_t* Bblk = Bt + (size_t)nblk * K;

  v8f zero = {0.f, 0.f, 0.f, 0.f, 0.f, 0.f, 0.f, 0.f};
  v8f acc[2][4];
#pragma unroll
  for (int i = 0; i < 2; ++i)
#pragma unroll
    for (int j = 0; j < 4; ++j) acc[i][j] = zero;

  if (threadIdx.x < 32)
    tdm_load_b_tile((unsigned)(uintptr_t)&smemB[0][0], Bblk, K);

  int buf = 0;
  for (int kep = 0; kep < K; kep += 64) {   // one 128x64 TDM tile per epoch
    if (threadIdx.x < 32) {
      if (kep + 64 < K) {
        tdm_load_b_tile((unsigned)(uintptr_t)&smemB[buf ^ 1][0], Bblk + kep + 64, K);
        __builtin_amdgcn_s_wait_tensorcnt(1);  // oldest DMA (this epoch) done
      } else {
        __builtin_amdgcn_s_wait_tensorcnt(0);
      }
    }
    __syncthreads();

    if (kep + 128 < K) {  // prefetch A two epochs ahead -> global_prefetch_b8
      __builtin_prefetch(A + (size_t)(m0 + (l & 15)) * K + kep + 128 + ((l >> 4) << 3), 0, 1);
      __builtin_prefetch(A + (size_t)(m0 + 16 + (l & 15)) * K + kep + 128 + ((l >> 4) << 3), 0, 1);
    }

    const bf16_t* Bl = &smemB[buf][0];
#pragma unroll
    for (int ks = 0; ks < 2; ++ks) {        // two 32-wide k-substeps
      int k0 = kep + ks * 32;
      v16bf a0 = load_a_frag(A, m0, k0, K);
      v16bf a1 = load_a_frag(A, m0 + 16, k0, K);
      v16bf bfr[4];
#pragma unroll
      for (int j = 0; j < 4; ++j)
        bfr[j] = load_b_frag_lds(Bl, n0l + 16 * j, ks);
#pragma unroll
      for (int j = 0; j < 4; ++j) {
        acc[0][j] = WMMA_BF16(a0, bfr[j], acc[0][j]);
        acc[1][j] = WMMA_BF16(a1, bfr[j], acc[1][j]);
      }
    }
    __syncthreads();
    buf ^= 1;
  }

  int rbase = (l >> 4) << 3;  // +0 or +8 row offset inside tile
  int nlane = l & 15;
#pragma unroll
  for (int i = 0; i < 2; ++i) {
#pragma unroll
    for (int j = 0; j < 4; ++j) {
#pragma unroll
      for (int r = 0; r < 8; ++r) {
        int m = m0 + 16 * i + r + rbase;
        int n = nblk + n0l + 16 * j + nlane;
        float v = acc[i][j][r];
        if (bias) v += bias[n];
        if (relu) v = fmaxf(v, 0.f);
        if (resid) v += resid[(size_t)m * N + n];
        if (transV) {
          int b = m >> 8, t = m & 255;
          int h = n >> 6, d = n & 63;
          outB[(((size_t)(b * 8 + h) * 64 + d) << 8) + t] = (bf16_t)v;
        } else {
          if (outF) outF[(size_t)m * N + n] = v;
          if (outB) outB[(size_t)m * N + n] = (bf16_t)v;
        }
      }
    }
  }
}

// ---------------------------------------------------------------------------
// Attention scores: P = softmax(mask(Q K^T * scale)), bf16 out [B,H,T,T].
// One block per (b, h, 32-query chunk): grid.x = 64*8*8 = 4096.
__global__ __launch_bounds__(256) void attn_scores_kernel(
    const bf16_t* __restrict__ Q, const bf16_t* __restrict__ Kmat,
    const int* __restrict__ mask, bf16_t* __restrict__ P, int causal) {
  int idx = blockIdx.x;
  int qb = idx & 7;
  int h  = (idx >> 3) & 7;
  int b  = idx >> 6;

  const bf16_t* Qb = Q    + (size_t)(b * 256 + qb * 32) * 512 + h * 64;
  const bf16_t* Kb = Kmat + (size_t)(b * 256) * 512 + h * 64;

  __shared__ float smem[32][256];
  __shared__ float red[32][8];

  int w = threadIdx.x >> 5;
  int l = threadIdx.x & 31;
  int mt  = w >> 2;         // 0..1 : query tile
  int ktg = (w & 3) * 4;    // key tile group base (4 tiles each)

  v8f zero = {0.f, 0.f, 0.f, 0.f, 0.f, 0.f, 0.f, 0.f};
  v8f acc[4] = {zero, zero, zero, zero};
#pragma unroll
  for (int k0 = 0; k0 < 64; k0 += 32) {
    v16bf a = load_a_frag(Qb, mt * 16, k0, 512);
#pragma unroll
    for (int j = 0; j < 4; ++j) {
      v16bf bf = load_b_frag(Kb, (ktg + j) * 16, k0, 512);
      acc[j] = WMMA_BF16(a, bf, acc[j]);
    }
  }
  int rbase = (l >> 4) << 3;
#pragma unroll
  for (int j = 0; j < 4; ++j)
#pragma unroll
    for (int r = 0; r < 8; ++r)
      smem[mt * 16 + r + rbase][(ktg + j) * 16 + (l & 15)] = acc[j][r] * SCALE_QK;
  __syncthreads();

  // Softmax: 8 threads per row, 32 cols each.
  int row = threadIdx.x >> 3;
  int c8  = threadIdx.x & 7;
  int q   = qb * 32 + row;
  const int* mrow = mask + ((size_t)b * 256 + q) * 256 + c8 * 32;

  float vals[32];
  float lmax = -3.0e38f;
#pragma unroll
  for (int i = 0; i < 32; ++i) {
    int col = c8 * 32 + i;
    bool ok = (mrow[i] != 0) && (!causal || col <= q);
    float v = ok ? smem[row][col] : -1e9f;
    vals[i] = v;
    lmax = fmaxf(lmax, v);
  }
  red[row][c8] = lmax;
  __syncthreads();
  float rmax = red[row][0];
#pragma unroll
  for (int i = 1; i < 8; ++i) rmax = fmaxf(rmax, red[row][i]);

  float lsum = 0.f;
#pragma unroll
  for (int i = 0; i < 32; ++i) {
    vals[i] = __expf(vals[i] - rmax);
    lsum += vals[i];
  }
  __syncthreads();
  red[row][c8] = lsum;
  __syncthreads();
  float rsum = 0.f;
#pragma unroll
  for (int i = 0; i < 8; ++i) rsum += red[row][i];
  float inv = 1.f / rsum;

  bf16_t* prow = P + ((size_t)(b * 8 + h) * 256 + q) * 256 + c8 * 32;
#pragma unroll
  for (int i = 0; i < 32; ++i) prow[i] = (bf16_t)(vals[i] * inv);
}

// ---------------------------------------------------------------------------
// out[b,t,h*64+d] = P[b,h] @ V[b,h]  (via vt[b,h,d,t] as B-operand).
// One block per (b,h): grid.x = 512. Wave w computes rows 32w..32w+31, d 0..63.
__global__ __launch_bounds__(256) void attn_pv_kernel(
    const bf16_t* __restrict__ P, const bf16_t* __restrict__ Vt,
    bf16_t* __restrict__ O) {
  int b = blockIdx.x >> 3;
  int h = blockIdx.x & 7;
  const bf16_t* Pb = P  + (size_t)(b * 8 + h) * 256 * 256;
  const bf16_t* Vb = Vt + (size_t)(b * 8 + h) * 64 * 256;

  int w = threadIdx.x >> 5;
  int l = threadIdx.x & 31;
  int m0 = w * 32;

  v8f zero = {0.f, 0.f, 0.f, 0.f, 0.f, 0.f, 0.f, 0.f};
  v8f acc[2][4];
#pragma unroll
  for (int i = 0; i < 2; ++i)
#pragma unroll
    for (int j = 0; j < 4; ++j) acc[i][j] = zero;

#pragma unroll 2
  for (int k0 = 0; k0 < 256; k0 += 32) {
    v16bf a0 = load_a_frag(Pb, m0, k0, 256);
    v16bf a1 = load_a_frag(Pb, m0 + 16, k0, 256);
#pragma unroll
    for (int j = 0; j < 4; ++j) {
      v16bf bf = load_b_frag(Vb, j * 16, k0, 256);
      acc[0][j] = WMMA_BF16(a0, bf, acc[0][j]);
      acc[1][j] = WMMA_BF16(a1, bf, acc[1][j]);
    }
  }
  int rbase = (l >> 4) << 3;
#pragma unroll
  for (int i = 0; i < 2; ++i)
#pragma unroll
    for (int j = 0; j < 4; ++j)
#pragma unroll
      for (int r = 0; r < 8; ++r) {
        int m = m0 + 16 * i + r + rbase;
        int n = 16 * j + (l & 15);
        O[(size_t)(b * 256 + m) * 512 + h * 64 + n] = (bf16_t)acc[i][j][r];
      }
}

// ---------------------------------------------------------------------------
extern "C" void kernel_launch(void* const* d_in, const int* in_sizes, int n_in,
                              void* d_out, int out_size, void* d_ws, size_t ws_size,
                              hipStream_t stream) {
  const float* x    = (const float*)d_in[0];
  const float* ca   = (const float*)d_in[1];
  const int*   x_m  = (const int*)d_in[2];
  const int*   ca_m = (const int*)d_in[3];
  const float* Wq_s = (const float*)d_in[4];
  const float* Wk_s = (const float*)d_in[5];
  const float* Wv_s = (const float*)d_in[6];
  const float* Wo_s = (const float*)d_in[7];
  const float* bo_s = (const float*)d_in[8];
  const float* Wq_c = (const float*)d_in[9];
  const float* Wk_c = (const float*)d_in[10];
  const float* Wv_c = (const float*)d_in[11];
  const float* Wo_c = (const float*)d_in[12];
  const float* bo_c = (const float*)d_in[13];
  const float* g1   = (const float*)d_in[14];
  const float* be1  = (const float*)d_in[15];
  const float* g2   = (const float*)d_in[16];
  const float* be2  = (const float*)d_in[17];
  const float* g3   = (const float*)d_in[18];
  const float* be3  = (const float*)d_in[19];
  const float* Wf1  = (const float*)d_in[20];
  const float* bf1  = (const float*)d_in[21];
  const float* Wf2  = (const float*)d_in[22];
  const float* bf2  = (const float*)d_in[23];

  const size_t MTOT = 16384;   // B*T
  char* ws = (char*)d_ws;
  size_t off = 0;
  auto alloc = [&](size_t bytes) -> void* {
    void* p = ws + off;
    off += (bytes + 255) & ~(size_t)255;
    return p;
  };
  float*  x_work = (float*)alloc(MTOT * 512 * 4);
  bf16_t* ln_bf  = (bf16_t*)alloc(MTOT * 512 * 2);   // aliased as attn output
  bf16_t* attn   = ln_bf;
  bf16_t* q_bf   = (bf16_t*)alloc(MTOT * 512 * 2);
  bf16_t* k_bf   = (bf16_t*)alloc(MTOT * 512 * 2);
  bf16_t* vt_bf  = (bf16_t*)alloc(MTOT * 512 * 2);   // [B,H,64,256]
  bf16_t* ca_bf  = (bf16_t*)alloc(MTOT * 512 * 2);
  bf16_t* p_bf   = (bf16_t*)alloc((size_t)64 * 8 * 256 * 256 * 2);  // also FFN hidden
  bf16_t* h1_bf  = p_bf;
  bf16_t* WqsT = (bf16_t*)alloc(512 * 512 * 2);
  bf16_t* WksT = (bf16_t*)alloc(512 * 512 * 2);
  bf16_t* WvsT = (bf16_t*)alloc(512 * 512 * 2);
  bf16_t* WosT = (bf16_t*)alloc(512 * 512 * 2);
  bf16_t* WqcT = (bf16_t*)alloc(512 * 512 * 2);
  bf16_t* WkcT = (bf16_t*)alloc(512 * 512 * 2);
  bf16_t* WvcT = (bf16_t*)alloc(512 * 512 * 2);
  bf16_t* WocT = (bf16_t*)alloc(512 * 512 * 2);
  bf16_t* Wf1T = (bf16_t*)alloc((size_t)512 * 2048 * 2);
  bf16_t* Wf2T = (bf16_t*)alloc((size_t)2048 * 512 * 2);

  // ---- weight prep -------------------------------------------------------
  wtrans_kernel<<<1024, 256, 0, stream>>>(Wq_s, WqsT, 512, 512);
  wtrans_kernel<<<1024, 256, 0, stream>>>(Wk_s, WksT, 512, 512);
  wtrans_kernel<<<1024, 256, 0, stream>>>(Wv_s, WvsT, 512, 512);
  wtrans_kernel<<<1024, 256, 0, stream>>>(Wo_s, WosT, 512, 512);
  wtrans_kernel<<<1024, 256, 0, stream>>>(Wq_c, WqcT, 512, 512);
  wtrans_kernel<<<1024, 256, 0, stream>>>(Wk_c, WkcT, 512, 512);
  wtrans_kernel<<<1024, 256, 0, stream>>>(Wv_c, WvcT, 512, 512);
  wtrans_kernel<<<1024, 256, 0, stream>>>(Wo_c, WocT, 512, 512);
  wtrans_kernel<<<4096, 256, 0, stream>>>(Wf1, Wf1T, 512, 2048);
  wtrans_kernel<<<4096, 256, 0, stream>>>(Wf2, Wf2T, 2048, 512);
  cvt_kernel<<<32768, 256, 0, stream>>>(ca, ca_bf, (int)(MTOT * 512));

  dim3 g512(4, 128);    // N=512 GEMMs
  dim3 g2048(16, 128);  // N=2048 GEMM

  // ---- self-attention ----------------------------------------------------
  ln_kernel<<<2048, 256, 0, stream>>>(x, g1, be1, ln_bf);
  gemm_bf16_kernel<<<g512, 256, 0, stream>>>(ln_bf, WqsT, 16384, 512, 512,
      nullptr, nullptr, nullptr, q_bf, 0, 0);
  gemm_bf16_kernel<<<g512, 256, 0, stream>>>(ln_bf, WksT, 16384, 512, 512,
      nullptr, nullptr, nullptr, k_bf, 0, 0);
  gemm_bf16_kernel<<<g512, 256, 0, stream>>>(ln_bf, WvsT, 16384, 512, 512,
      nullptr, nullptr, nullptr, vt_bf, 0, 1);
  attn_scores_kernel<<<4096, 256, 0, stream>>>(q_bf, k_bf, x_m, p_bf, 1);
  attn_pv_kernel<<<512, 256, 0, stream>>>(p_bf, vt_bf, attn);
  gemm_bf16_kernel<<<g512, 256, 0, stream>>>(attn, WosT, 16384, 512, 512,
      bo_s, x, x_work, nullptr, 0, 0);

  // ---- cross-attention ---------------------------------------------------
  ln_kernel<<<2048, 256, 0, stream>>>(x_work, g2, be2, ln_bf);
  gemm_bf16_kernel<<<g512, 256, 0, stream>>>(ln_bf, WqcT, 16384, 512, 512,
      nullptr, nullptr, nullptr, q_bf, 0, 0);
  gemm_bf16_kernel<<<g512, 256, 0, stream>>>(ca_bf, WkcT, 16384, 512, 512,
      nullptr, nullptr, nullptr, k_bf, 0, 0);
  gemm_bf16_kernel<<<g512, 256, 0, stream>>>(ca_bf, WvcT, 16384, 512, 512,
      nullptr, nullptr, nullptr, vt_bf, 0, 1);
  attn_scores_kernel<<<4096, 256, 0, stream>>>(q_bf, k_bf, ca_m, p_bf, 0);
  attn_pv_kernel<<<512, 256, 0, stream>>>(p_bf, vt_bf, attn);
  gemm_bf16_kernel<<<g512, 256, 0, stream>>>(attn, WocT, 16384, 512, 512,
      bo_c, x_work, x_work, nullptr, 0, 0);

  // ---- feed-forward ------------------------------------------------------
  ln_kernel<<<2048, 256, 0, stream>>>(x_work, g3, be3, ln_bf);
  gemm_bf16_kernel<<<g2048, 256, 0, stream>>>(ln_bf, Wf1T, 16384, 2048, 512,
      bf1, nullptr, nullptr, h1_bf, 1, 0);
  gemm_bf16_kernel<<<g512, 256, 0, stream>>>(h1_bf, Wf2T, 16384, 512, 2048,
      bf2, x_work, (float*)d_out, nullptr, 0, 0);
}
